// StreamProcessor_79010218377433
// MI455X (gfx1250) — matrix operand order repaired and checked by reading, not compiled, not run on hardware
//
#include <hip/hip_runtime.h>
#include <hip/hip_bf16.h>

// ---- model constants ----
#define B_SZ    16
#define L_SEQ   3600
#define LATENT  64
#define D_MODEL 128
#define D_INNER 256
#define D_STATE 16
#define D_CONV  4
#define DT_RANK 8
#define M_ROWS  (B_SZ * L_SEQ)          // 57600, multiple of 16
#define N_XZ    (2 * D_INNER)           // 512
#define N_DBC   (D_INNER + 2 * D_STATE) // 288

typedef _Float16 h16;
typedef __attribute__((ext_vector_type(16))) _Float16     v16h;
typedef __attribute__((ext_vector_type(8)))  float        v8f;
typedef __attribute__((ext_vector_type(4)))  unsigned int v4u;

union Frag16 { v16h h; v4u u[2]; };

// ---------------------------------------------------------------------------
// Per-wave 16x16 WMMA tile:  C(16x16,f32) = A(16xK,f16 rowmajor) * W(NxK,f16
// rowmajor)^T   (K multiple of 32).  Fragment layouts per CDNA5 ISA 7.12.2.
// ---------------------------------------------------------------------------
__device__ __forceinline__ v8f wmma_tile_f16(const h16* __restrict__ A,
                                             const h16* __restrict__ W,
                                             int K, int m0, int n0, int lane) {
  const int l16 = lane & 15;
  const int hf  = lane >> 4;
  v8f c = {0.f, 0.f, 0.f, 0.f, 0.f, 0.f, 0.f, 0.f};
  const h16* arow = A + (size_t)(m0 + l16) * K;
  const h16* wrow = W + (size_t)(n0 + l16) * K;
  for (int k0 = 0; k0 < K; k0 += 32) {
    Frag16 fa, fb;
    // A 16x32: lanes 0-15 hold K=[0..7,16..23], lanes 16-31 hold K=[8..15,24..31]
    fa.u[0] = *(const v4u*)(arow + k0 + 8 * hf);        // v0..v3
    fa.u[1] = *(const v4u*)(arow + k0 + 16 + 8 * hf);   // v4..v7
    // B 32x16: N = l16, K order same as A
    fb.u[0] = *(const v4u*)(wrow + k0 + 8 * hf);
    fb.u[1] = *(const v4u*)(wrow + k0 + 16 + 8 * hf);
    c = __builtin_amdgcn_wmma_f32_16x16x32_f16(false, fa.h, false, fb.h,
                                               (short)0, c, false, false);
  }
  return c;
}

// ---------------------------------------------------------------------------
// Weight prep: W1 = in_proj_w @ emb_w  (512x64), b1 = in_proj_w @ emb_b
// ---------------------------------------------------------------------------
__global__ void k_prep_w1(const float* __restrict__ inw,
                          const float* __restrict__ embw,
                          const float* __restrict__ embb,
                          h16* __restrict__ W1, float* __restrict__ b1) {
  int idx = blockIdx.x * blockDim.x + threadIdx.x;
  if (idx >= N_XZ * LATENT) return;
  int o = idx / LATENT, i = idx % LATENT;
  float s = 0.f;
  for (int j = 0; j < D_MODEL; ++j) s += inw[o * D_MODEL + j] * embw[j * LATENT + i];
  W1[idx] = (h16)s;
  if (i == 0) {
    float sb = 0.f;
    for (int j = 0; j < D_MODEL; ++j) sb += inw[o * D_MODEL + j] * embb[j];
    b1[o] = sb;
  }
}

// W2 rows 0..255 = dt_proj_w @ x_proj_w[0:8]; rows 256..287 = x_proj_w[8:40]
__global__ void k_prep_w2(const float* __restrict__ dtw,
                          const float* __restrict__ xpw,
                          h16* __restrict__ W2) {
  int idx = blockIdx.x * blockDim.x + threadIdx.x;
  if (idx >= N_DBC * D_INNER) return;
  int row = idx / D_INNER, i = idx % D_INNER;
  float v;
  if (row < D_INNER) {
    v = 0.f;
    for (int r = 0; r < DT_RANK; ++r) v += dtw[row * DT_RANK + r] * xpw[r * D_INNER + i];
  } else {
    v = xpw[(DT_RANK + row - D_INNER) * D_INNER + i];
  }
  W2[idx] = (h16)v;
}

__global__ void k_f32_to_f16(const float* __restrict__ src, h16* __restrict__ dst, int n) {
  int i = blockIdx.x * blockDim.x + threadIdx.x;
  if (i < n) dst[i] = (h16)src[i];
}

// ---------------------------------------------------------------------------
// GEMM1: xz = z_f16(57600x64) @ W1^T(64x512) + b1.  Store x half as f16
// (conv input); z half only needed at t = L-1 -> store silu(z) there.
// ---------------------------------------------------------------------------
__global__ void k_gemm_xz(const h16* __restrict__ zf16, const h16* __restrict__ W1,
                          const float* __restrict__ b1,
                          h16* __restrict__ x_pre, float* __restrict__ z_last) {
  int wave = (blockIdx.x * blockDim.x + threadIdx.x) >> 5;
  int lane = threadIdx.x & 31;
  const int nTiles = N_XZ / 16;  // 32
  int m0 = (wave / nTiles) * 16;
  int n0 = (wave % nTiles) * 16;
  if (m0 >= M_ROWS) return;
  v8f c = wmma_tile_f16(zf16, W1, LATENT, m0, n0, lane);
  int l16 = lane & 15, hf = lane >> 4;
  int n = n0 + l16;
  float bias = b1[n];
  #pragma unroll
  for (int r = 0; r < 8; ++r) {
    int mm = m0 + r + 8 * hf;            // C/D layout: lanes16-31 -> M=8..15
    float v = c[r] + bias;
    if (n < D_INNER) {
      x_pre[(size_t)mm * D_INNER + n] = (h16)v;
    } else {
      if ((mm % L_SEQ) == L_SEQ - 1) {
        float sg = 1.f / (1.f + __expf(-v));
        z_last[(mm / L_SEQ) * D_INNER + (n - D_INNER)] = v * sg;
      }
    }
  }
}

// ---------------------------------------------------------------------------
// Depthwise causal conv (width 4) + SiLU, f16 in/out
// ---------------------------------------------------------------------------
__global__ void k_conv(const h16* __restrict__ x_pre, const float* __restrict__ cw,
                       const float* __restrict__ cb, h16* __restrict__ xs) {
  int idx = blockIdx.x * blockDim.x + threadIdx.x;   // 57600*256 threads
  int d = idx & (D_INNER - 1);
  int m = idx >> 8;
  int t = m % L_SEQ;
  float sum = cb[d];
  #pragma unroll
  for (int j = 0; j < D_CONV; ++j) {
    int tt = t - (D_CONV - 1) + j;
    if (tt >= 0)
      sum += cw[d * D_CONV + j] * (float)x_pre[idx + (j - (D_CONV - 1)) * D_INNER];
  }
  float sg = 1.f / (1.f + __expf(-sum));
  xs[idx] = (h16)(sum * sg);
}

// ---------------------------------------------------------------------------
// GEMM2: [dt_lin | B | C] = x_silu(57600x256) @ W2^T(256x288)
// Epilogue: softplus(dt_lin + dt_proj_b) -> dt (f32); split B/C (f32)
// ---------------------------------------------------------------------------
__global__ void k_gemm_dbc(const h16* __restrict__ xs, const h16* __restrict__ W2,
                           const float* __restrict__ dtb,
                           float* __restrict__ dt, float* __restrict__ Bc,
                           float* __restrict__ Cc) {
  int wave = (blockIdx.x * blockDim.x + threadIdx.x) >> 5;
  int lane = threadIdx.x & 31;
  const int nTiles = N_DBC / 16;  // 18
  int m0 = (wave / nTiles) * 16;
  int n0 = (wave % nTiles) * 16;
  if (m0 >= M_ROWS) return;
  v8f c = wmma_tile_f16(xs, W2, D_INNER, m0, n0, lane);
  int l16 = lane & 15, hf = lane >> 4;
  int n = n0 + l16;
  #pragma unroll
  for (int r = 0; r < 8; ++r) {
    int mm = m0 + r + 8 * hf;
    float v = c[r];
    if (n < D_INNER) {
      float u = v + dtb[n];
      float sp = (u > 20.f) ? u : __logf(1.f + __expf(u));   // softplus
      dt[(size_t)mm * D_INNER + n] = sp;
    } else if (n < D_INNER + D_STATE) {
      Bc[(size_t)mm * D_STATE + (n - D_INNER)] = v;
    } else {
      Cc[(size_t)mm * D_STATE + (n - D_INNER - D_STATE)] = v;
    }
  }
}

// ---------------------------------------------------------------------------
// Selective scan: 4096 independent (b,d) recurrences, lane = (pair, state s).
// Only the final-step y is needed, so no per-step reduction.
// ---------------------------------------------------------------------------
__global__ void k_scan(const float* __restrict__ dt, const h16* __restrict__ xs,
                       const float* __restrict__ Bc, const float* __restrict__ Cc,
                       const float* __restrict__ A_log, const float* __restrict__ Dsk,
                       const float* __restrict__ z_last, float* __restrict__ yfin) {
  int g = blockIdx.x * blockDim.x + threadIdx.x;
  int lane = g & 31;
  int wave = g >> 5;
  int pair = wave * 2 + (lane >> 4);    // (b,d) pair; 2 per wave
  int s = lane & 15;                    // state index
  int b = pair >> 8;
  int d = pair & (D_INNER - 1);
  float a = -__expf(A_log[d * D_STATE + s]);
  float dskip = Dsk[d];
  float h = 0.f, xv = 0.f;
  const size_t rowbase = (size_t)b * L_SEQ;
  for (int t = 0; t < L_SEQ; ++t) {
    size_t idx = rowbase + t;
    float dtv = dt[idx * D_INNER + d];
    xv = (float)xs[idx * D_INNER + d];
    float bv = Bc[idx * D_STATE + s];
    h = __expf(dtv * a) * h + (dtv * xv) * bv;
    __builtin_prefetch(&dt[(idx + 16) * D_INNER + d], 0, 1);  // global_prefetch_b8
  }
  size_t last = rowbase + L_SEQ - 1;
  float part = h * Cc[last * D_STATE + s];
  part += __shfl_xor(part, 8);
  part += __shfl_xor(part, 4);
  part += __shfl_xor(part, 2);
  part += __shfl_xor(part, 1);
  if (s == 0) {
    float y = part + xv * dskip;                 // xv == x_silu at t=L-1
    yfin[b * D_INNER + d] = y * z_last[b * D_INNER + d];
  }
}

// ---------------------------------------------------------------------------
// Head: out_last = yfin @ out_proj_w^T, layernorm(128), classifier -> (B,)
// ---------------------------------------------------------------------------
__global__ void k_head(const float* __restrict__ yfin, const float* __restrict__ outw,
                       const float* __restrict__ nw, const float* __restrict__ nb,
                       const float* __restrict__ clsw, const float* __restrict__ clsb,
                       float* __restrict__ out) {
  __shared__ float sy[D_INNER];
  __shared__ float red[D_MODEL];
  int b = blockIdx.x, n = threadIdx.x;    // 128 threads
  sy[n]           = yfin[b * D_INNER + n];
  sy[n + D_MODEL] = yfin[b * D_INNER + n + D_MODEL];
  __syncthreads();
  float acc = 0.f;
  for (int i = 0; i < D_INNER; ++i) acc += sy[i] * outw[n * D_INNER + i];
  red[n] = acc; __syncthreads();
  for (int off = 64; off; off >>= 1) { if (n < off) red[n] += red[n + off]; __syncthreads(); }
  float mu = red[0] / (float)D_MODEL; __syncthreads();
  float dv = acc - mu;
  red[n] = dv * dv; __syncthreads();
  for (int off = 64; off; off >>= 1) { if (n < off) red[n] += red[n + off]; __syncthreads(); }
  float var = red[0] / (float)D_MODEL; __syncthreads();
  float xn = dv * rsqrtf(var + 1e-5f) * nw[n] + nb[n];
  red[n] = xn * clsw[n]; __syncthreads();
  for (int off = 64; off; off >>= 1) { if (n < off) red[n] += red[n + off]; __syncthreads(); }
  if (n == 0) out[b] = red[0] + clsb[0];
}

// ---------------------------------------------------------------------------
extern "C" void kernel_launch(void* const* d_in, const int* in_sizes, int n_in,
                              void* d_out, int out_size, void* d_ws, size_t ws_size,
                              hipStream_t stream) {
  const float* z_seq      = (const float*)d_in[0];
  const float* emb_w      = (const float*)d_in[1];
  const float* emb_b      = (const float*)d_in[2];
  const float* in_proj_w  = (const float*)d_in[3];
  const float* conv_w     = (const float*)d_in[4];
  const float* conv_b     = (const float*)d_in[5];
  const float* x_proj_w   = (const float*)d_in[6];
  const float* dt_proj_w  = (const float*)d_in[7];
  const float* dt_proj_b  = (const float*)d_in[8];
  const float* A_log      = (const float*)d_in[9];
  const float* D_skip     = (const float*)d_in[10];
  const float* out_proj_w = (const float*)d_in[11];
  const float* norm_w     = (const float*)d_in[12];
  const float* norm_b     = (const float*)d_in[13];
  const float* cls_w      = (const float*)d_in[14];
  const float* cls_b      = (const float*)d_in[15];
  float* out = (float*)d_out;

  // workspace carve (256B aligned)
  char* p = (char*)d_ws;
  auto carve = [&](size_t bytes) -> void* {
    void* r = (void*)p;
    p += (bytes + 255) & ~(size_t)255;
    return r;
  };
  h16*   zf16   = (h16*)  carve((size_t)M_ROWS * LATENT * 2);
  h16*   W1     = (h16*)  carve((size_t)N_XZ * LATENT * 2);
  float* b1     = (float*)carve((size_t)N_XZ * 4);
  h16*   W2     = (h16*)  carve((size_t)N_DBC * D_INNER * 2);
  h16*   x_pre  = (h16*)  carve((size_t)M_ROWS * D_INNER * 2);
  h16*   x_silu = (h16*)  carve((size_t)M_ROWS * D_INNER * 2);
  float* dt     = (float*)carve((size_t)M_ROWS * D_INNER * 4);
  float* Bc     = (float*)carve((size_t)M_ROWS * D_STATE * 4);
  float* Cc     = (float*)carve((size_t)M_ROWS * D_STATE * 4);
  float* z_last = (float*)carve((size_t)B_SZ * D_INNER * 4);
  float* yfin   = (float*)carve((size_t)B_SZ * D_INNER * 4);

  // 0) convert z_sequence to f16 (WMMA A operand)
  {
    int n = M_ROWS * LATENT;
    k_f32_to_f16<<<(n + 255) / 256, 256, 0, stream>>>(z_seq, zf16, n);
  }
  // 1) weight fusions
  k_prep_w1<<<(N_XZ * LATENT + 255) / 256, 256, 0, stream>>>(in_proj_w, emb_w, emb_b, W1, b1);
  k_prep_w2<<<(N_DBC * D_INNER + 255) / 256, 256, 0, stream>>>(dt_proj_w, x_proj_w, W2);
  // 2) fused emb+in_proj GEMM (WMMA)
  {
    int tiles = (M_ROWS / 16) * (N_XZ / 16);       // 115200
    k_gemm_xz<<<tiles / 8, 256, 0, stream>>>(zf16, W1, b1, x_pre, z_last);
  }
  // 3) depthwise causal conv + SiLU
  k_conv<<<(M_ROWS * D_INNER) / 256, 256, 0, stream>>>(x_pre, conv_w, conv_b, x_silu);
  // 4) fused x_proj + dt_proj GEMM (WMMA) -> dt (softplus), B, C
  {
    int tiles = (M_ROWS / 16) * (N_DBC / 16);      // 64800
    k_gemm_dbc<<<tiles / 8, 256, 0, stream>>>(x_silu, W2, dt_proj_b, dt, Bc, Cc);
  }
  // 5) selective scan (lane-per-state, 2 (b,d) pairs per wave)
  k_scan<<<(B_SZ * D_INNER * D_STATE) / 256, 256, 0, stream>>>(
      dt, x_silu, Bc, Cc, A_log, D_skip, z_last, yfin);
  // 6) head: out-proj(last) + layernorm + classifier
  k_head<<<B_SZ, D_MODEL, 0, stream>>>(yfin, out_proj_w, norm_w, norm_b, cls_w, cls_b, out);

  (void)in_sizes; (void)n_in; (void)out_size; (void)ws_size;
}
